// W8A16Linear_28810640621893
// MI455X (gfx1250) — matrix operand order repaired and checked by reading, not compiled, not run on hardware
//
#include <hip/hip_runtime.h>

typedef __attribute__((ext_vector_type(16))) _Float16 v16h;
typedef __attribute__((ext_vector_type(8)))  _Float16 v8h;
typedef __attribute__((ext_vector_type(8)))  float    v8f;
typedef __attribute__((ext_vector_type(4)))  int      v4i;

#define IN_F   4096
#define OUT_F  16384
#define TOKENS 64

// One wave owns 16 output channels (one N-tile) and all 64 tokens (4 M-tiles).
// Weight is streamed exactly once (non-temporal), x is reused from cache by
// every wave, per-channel scale/bias applied once in the epilogue.
__global__ __launch_bounds__(128, 1) void w8a16_wmma_kernel(
    const _Float16* __restrict__ x,       // [64, 4096] fp16
    const int*      __restrict__ weight,  // [16384, 4096] int8-in-int32
    const _Float16* __restrict__ scales,  // [16384] fp16
    const _Float16* __restrict__ bias,    // [16384] fp16
    float*          __restrict__ out)     // [64, 16384] f32
{
    const int lane = threadIdx.x & 31;
    const int wave = threadIdx.x >> 5;
    const int col  = lane & 15;   // N index within tile / M row for A loads
    const int hi   = lane >> 4;   // upper half-wave: K phase selector

    const int ntile = blockIdx.x * 4 + wave;   // 0..1023
    const int nBase = ntile * 16;

    // B source: this lane streams weight row (output channel) nBase+col.
    // lanes<16 take K = k..k+15, lanes>=16 take K = k+16..k+31.
    const int* wrow = weight + (size_t)(nBase + col) * IN_F + hi * 16;

    // A source: row M = col of each 16-token tile.
    // lanes<16: K = k..k+15; lanes>=16: K = k+16..k+31.
    const _Float16* xbase = x + (size_t)col * IN_F + hi * 16;

    v8f acc0 = {}, acc1 = {}, acc2 = {}, acc3 = {};

    for (int k = 0; k < IN_F; k += 32) {
        // ---- B tile: 32x16 weights, non-temporal stream + prefetch-ahead ----
        v4i w0 = __builtin_nontemporal_load((const v4i*)(wrow + k + 0));
        v4i w1 = __builtin_nontemporal_load((const v4i*)(wrow + k + 4));
        v4i w2 = __builtin_nontemporal_load((const v4i*)(wrow + k + 8));
        v4i w3 = __builtin_nontemporal_load((const v4i*)(wrow + k + 12));
        __builtin_prefetch(wrow + k + 512, 0, 1);   // ~2KB ahead in this row

        v16h b;
        #pragma unroll
        for (int j = 0; j < 4; ++j) {
            b[j]      = (_Float16)w0[j];
            b[4 + j]  = (_Float16)w1[j];
            b[8 + j]  = (_Float16)w2[j];
            b[12 + j] = (_Float16)w3[j];
        }

        // ---- 4 M-tiles of A, one WMMA each ----
        const _Float16* xk = xbase + k;
        #define MTILE(T, ACC) {                                                   \
            v8h a0 = *(const v8h*)(xk + (size_t)(T) * 16 * IN_F);                 \
            v8h a1 = *(const v8h*)(xk + (size_t)(T) * 16 * IN_F + 8);             \
            v16h a = __builtin_shufflevector(a0, a1,                              \
                         0,1,2,3,4,5,6,7,8,9,10,11,12,13,14,15);                  \
            ACC = __builtin_amdgcn_wmma_f32_16x16x32_f16(                         \
                      false, a, false, b, (short)0, ACC, false, false);           \
        }
        MTILE(0, acc0)
        MTILE(1, acc1)
        MTILE(2, acc2)
        MTILE(3, acc3)
        #undef MTILE
    }

    // ---- epilogue: per-channel scale & bias, f32 store ----
    const float s  = (float)scales[nBase + col];
    const float bb = (float)bias[nBase + col];

    // C/D layout: VGPR j holds M = j + 8*hi, lane col holds N.
    float* o = out + (size_t)(hi * 8) * OUT_F + nBase + col;
    #pragma unroll
    for (int j = 0; j < 8; ++j) {
        o[(size_t)(j)      * OUT_F] = acc0[j] * s + bb;  // tokens  0..15
        o[(size_t)(16 + j) * OUT_F] = acc1[j] * s + bb;  // tokens 16..31
        o[(size_t)(32 + j) * OUT_F] = acc2[j] * s + bb;  // tokens 32..47
        o[(size_t)(48 + j) * OUT_F] = acc3[j] * s + bb;  // tokens 48..63
    }
}

extern "C" void kernel_launch(void* const* d_in, const int* in_sizes, int n_in,
                              void* d_out, int out_size, void* d_ws, size_t ws_size,
                              hipStream_t stream) {
    const _Float16* x      = (const _Float16*)d_in[0];
    const int*      weight = (const int*)d_in[1];
    const _Float16* scales = (const _Float16*)d_in[2];
    const _Float16* bias   = (const _Float16*)d_in[3];
    float*          out    = (float*)d_out;

    dim3 grid(OUT_F / 16 / 4);   // 256 blocks, 4 waves each = 1024 N-tiles
    dim3 block(128);
    hipLaunchKernelGGL(w8a16_wmma_kernel, grid, block, 0, stream,
                       x, weight, scales, bias, out);
}
